// GNN_67963562492235
// MI455X (gfx1250) — compile-verified, matrix operand order checked by reading, not run on hardware
//
#include <hip/hip_runtime.h>
#include <float.h>

// CDNA5 / gfx1250 wave32 WMMA types
typedef __attribute__((ext_vector_type(16))) __bf16 v16bf;
typedef __attribute__((ext_vector_type(8)))  float  v8f;

#define FEAT 64
#define WLDS_STRIDE 68   // 64 + 4 pad: strided fragment reads hit disjoint bank ranges per lane-half

// ---------------------------------------------------------------- utilities

__global__ void fill_f32(float* __restrict__ p, float v, int n) {
    int i = blockIdx.x * blockDim.x + threadIdx.x;
    if (i < n) p[i] = v;
}

__global__ void deg_accum(const int* __restrict__ ei, float* __restrict__ deg, int E) {
    int e = blockIdx.x * blockDim.x + threadIdx.x;
    if (e < E) atomicAdd(&deg[ei[E + e]], 1.0f);   // dst row of edge_index
}

__global__ void rsqrt_inplace(float* __restrict__ d, int n) {
    int i = blockIdx.x * blockDim.x + threadIdx.x;
    if (i < n) d[i] = rsqrtf(d[i]);                // deg >= 1 always (self-loops)
}

__device__ inline void atomic_max_f32(float* addr, float val) {
    // monotone int/uint mapping trick; init value must be -FLT_MAX
    if (val >= 0.0f) atomicMax((int*)addr, __float_as_int(val));
    else             atomicMin((unsigned int*)addr, (unsigned int)__float_as_int(val));
}

// ------------------------------------------------- WMMA GEMM: C[N,64] = A[N,64] @ W[64,64]
// One wave -> 32 rows (2 M-tiles) x 64 cols (4 N-tiles); K=64 as 2 x v_wmma_f32_16x16x32_bf16.
// W is staged once per block into padded LDS (coalesced b128 loads), shared by all 4 waves.
// A-frag (16x32 bf16, ISA 7.12.2): lane holds row M=lane%16; K = base..base+7 and
//   16+base..16+base+7 with base = (lane/16)*8.  B-frag mirrors with N as lane index.
// C/D (16x16 f32): VGPR j -> M = j + (lane/16)*8, N = lane%16.
template <bool HAS_BIAS, bool SEGMAX>
__global__ __launch_bounds__(128) void gemm64_wmma(
    const float* __restrict__ A, const float* __restrict__ W,
    const float* __restrict__ bias, float* __restrict__ Out,
    const int* __restrict__ batch, int nrows)
{
    __shared__ float wlds[FEAT * WLDS_STRIDE];

    // ---- cooperative W stage: 4096 floats via 8 x b128 per thread, coalesced
    {
        const float4* W4 = (const float4*)W;
        #pragma unroll
        for (int it = 0; it < 8; ++it) {
            int i4 = threadIdx.x + it * 128;       // float4 index 0..1023
            float4 v = W4[i4];
            int i = i4 * 4;
            int k = i >> 6, n = i & 63;
            *(float4*)&wlds[k * WLDS_STRIDE + n] = v;
        }
    }
    __syncthreads();

    const int wave = threadIdx.x >> 5;
    const int lane = threadIdx.x & 31;
    const int row0 = (blockIdx.x * 4 + wave) * 32;
    if (row0 >= nrows) return;                 // wave-uniform exit (after barrier)
    const int half = lane >> 4;                // 0 | 1
    const int lm   = lane & 15;

    // ---- B fragments from LDS: 4 column tiles x 2 K-halves
    v16bf Bf[4][2];
    #pragma unroll
    for (int nt = 0; nt < 4; ++nt) {
        const int n = nt * 16 + lm;
        #pragma unroll
        for (int kh = 0; kh < 2; ++kh) {
            const int kb = kh * 32 + half * 8;
            v16bf b;
            #pragma unroll
            for (int j = 0; j < 8; ++j) {
                b[j]     = (__bf16)wlds[(kb + j)      * WLDS_STRIDE + n];
                b[8 + j] = (__bf16)wlds[(kb + 16 + j) * WLDS_STRIDE + n];
            }
            Bf[nt][kh] = b;
        }
    }

    // ---- A fragments: 2 M-tiles x 2 K-halves, b128 loads
    v16bf Af[2][2];
    #pragma unroll
    for (int mt = 0; mt < 2; ++mt) {
        int m = row0 + mt * 16 + lm;
        if (m >= nrows) m = nrows - 1;         // clamp: out-of-range tiles are never stored
        const float4* arow = (const float4*)(A + (size_t)m * FEAT);
        #pragma unroll
        for (int kh = 0; kh < 2; ++kh) {
            const int q = kh * 8 + half * 2;   // float4 index of k-base
            float4 p0 = arow[q];
            float4 p1 = arow[q + 1];
            float4 p2 = arow[q + 4];           // +16 floats
            float4 p3 = arow[q + 5];
            v16bf a;
            a[0]=(__bf16)p0.x; a[1]=(__bf16)p0.y; a[2]=(__bf16)p0.z; a[3]=(__bf16)p0.w;
            a[4]=(__bf16)p1.x; a[5]=(__bf16)p1.y; a[6]=(__bf16)p1.z; a[7]=(__bf16)p1.w;
            a[8]=(__bf16)p2.x; a[9]=(__bf16)p2.y; a[10]=(__bf16)p2.z; a[11]=(__bf16)p2.w;
            a[12]=(__bf16)p3.x; a[13]=(__bf16)p3.y; a[14]=(__bf16)p3.z; a[15]=(__bf16)p3.w;
            Af[mt][kh] = a;
        }
    }

    // ---- 16 WMMAs (EXEC all-1s here; branches above are wave-uniform)
    v8f acc[2][4];
    #pragma unroll
    for (int mt = 0; mt < 2; ++mt) {
        #pragma unroll
        for (int nt = 0; nt < 4; ++nt) {
            v8f c = {};
            c = __builtin_amdgcn_wmma_f32_16x16x32_bf16(false, Af[mt][0], false, Bf[nt][0],
                                                        (short)0, c, false, false);
            c = __builtin_amdgcn_wmma_f32_16x16x32_bf16(false, Af[mt][1], false, Bf[nt][1],
                                                        (short)0, c, false, false);
            acc[mt][nt] = c;
        }
    }

    // ---- bias hoist (compile-time selected, one load per column tile)
    float bvals[4] = {0.0f, 0.0f, 0.0f, 0.0f};
    if (HAS_BIAS) {
        #pragma unroll
        for (int nt = 0; nt < 4; ++nt) bvals[nt] = bias[nt * 16 + lm];
    }

    // ---- stores: wave-uniform full-tile fast path, guarded tail otherwise
    #pragma unroll
    for (int mt = 0; mt < 2; ++mt) {
        const int rbase = row0 + mt * 16;
        if (rbase >= nrows) continue;
        if (rbase + 16 <= nrows) {             // full tile: unconditional
            if (SEGMAX) {
                int bidx[8];
                #pragma unroll
                for (int j = 0; j < 8; ++j) bidx[j] = batch[rbase + j + half * 8];
                #pragma unroll
                for (int nt = 0; nt < 4; ++nt) {
                    const int n = nt * 16 + lm;
                    #pragma unroll
                    for (int j = 0; j < 8; ++j)
                        atomic_max_f32(&Out[(size_t)bidx[j] * FEAT + n],
                                       acc[mt][nt][j] + bvals[nt]);
                }
            } else {
                #pragma unroll
                for (int nt = 0; nt < 4; ++nt) {
                    const int n = nt * 16 + lm;
                    #pragma unroll
                    for (int j = 0; j < 8; ++j)
                        Out[(size_t)(rbase + j + half * 8) * FEAT + n] =
                            acc[mt][nt][j] + bvals[nt];
                }
            }
        } else {                               // partial tile (never taken at N%32==0)
            #pragma unroll
            for (int nt = 0; nt < 4; ++nt) {
                const int n = nt * 16 + lm;
                #pragma unroll
                for (int j = 0; j < 8; ++j) {
                    const int mo = rbase + j + half * 8;
                    if (mo < nrows) {
                        float v = acc[mt][nt][j] + bvals[nt];
                        if (SEGMAX) atomic_max_f32(&Out[(size_t)batch[mo] * FEAT + n], v);
                        else        Out[(size_t)mo * FEAT + n] = v;
                    }
                }
            }
        }
    }
}

// ------------------------------------------- message scatter: agg[dst] += lin[src]*norm
// 4 threads per message, 16 contiguous features each; explicit b128 gathers, f32 atomics.
__global__ void scatter_msgs(const float* __restrict__ lin, const int* __restrict__ ei,
                             const float* __restrict__ dinv, float* __restrict__ agg,
                             int E, int N)
{
    int tid = blockIdx.x * blockDim.x + threadIdx.x;
    int total = (E + N) * 4;
    if (tid >= total) return;
    int e = tid >> 2;
    int c = (tid & 3) * 16;
    int s, d;
    if (e < E) { s = ei[e]; d = ei[E + e]; }
    else       { s = d = e - E; }                       // self loop
    float norm = dinv[s] * dinv[d];
    const float4* srow = (const float4*)(lin + (size_t)s * FEAT + c);
    float*        drow = agg + (size_t)d * FEAT + c;
    float4 v0 = srow[0], v1 = srow[1], v2 = srow[2], v3 = srow[3];
    atomicAdd(drow +  0, v0.x * norm); atomicAdd(drow +  1, v0.y * norm);
    atomicAdd(drow +  2, v0.z * norm); atomicAdd(drow +  3, v0.w * norm);
    atomicAdd(drow +  4, v1.x * norm); atomicAdd(drow +  5, v1.y * norm);
    atomicAdd(drow +  6, v1.z * norm); atomicAdd(drow +  7, v1.w * norm);
    atomicAdd(drow +  8, v2.x * norm); atomicAdd(drow +  9, v2.y * norm);
    atomicAdd(drow + 10, v2.z * norm); atomicAdd(drow + 11, v2.w * norm);
    atomicAdd(drow + 12, v3.x * norm); atomicAdd(drow + 13, v3.y * norm);
    atomicAdd(drow + 14, v3.z * norm); atomicAdd(drow + 15, v3.w * norm);
}

// ------------------------------------------------------------- training-mode batchnorm
__global__ __launch_bounds__(256) void bn_stats(const float* __restrict__ h,
                                                float* __restrict__ stats, int N)
{
    __shared__ float ssum[256], ssq[256];      // [4 partials][64 features]
    int f = threadIdx.x & 63;
    int p = threadIdx.x >> 6;                  // 0..3
    int row0 = blockIdx.x * 256;
    int rend = row0 + 256; if (rend > N) rend = N;
    float s = 0.0f, q = 0.0f;
    for (int r = row0 + p; r < rend; r += 4) {
        float v = h[(size_t)r * FEAT + f];
        s += v; q += v * v;
    }
    ssum[p * 64 + f] = s; ssq[p * 64 + f] = q;
    __syncthreads();
    if (p == 0) {
        s = ssum[f] + ssum[64 + f] + ssum[128 + f] + ssum[192 + f];
        q = ssq [f] + ssq [64 + f] + ssq [128 + f] + ssq [192 + f];
        atomicAdd(&stats[f], s);
        atomicAdd(&stats[64 + f], q);
    }
}

__global__ void bn_apply_relu(const float* __restrict__ h, const float* __restrict__ stats,
                              const float* __restrict__ g, const float* __restrict__ b,
                              float* __restrict__ out, int N)
{
    int tid = blockIdx.x * blockDim.x + threadIdx.x;
    if (tid >= N * FEAT) return;
    int f = tid & 63;
    float inv_n = 1.0f / (float)N;
    float mu  = stats[f] * inv_n;
    float var = stats[64 + f] * inv_n - mu * mu;
    float rs  = rsqrtf(var + 1e-5f);
    float v   = (h[tid] - mu) * rs * g[f] + b[f];
    out[tid] = v > 0.0f ? v : 0.0f;
}

// --------------------------------------------------------- head: ge @ fc3 + log_softmax
__global__ void head_logsoftmax(const float* __restrict__ ge, const float* __restrict__ w3,
                                const float* __restrict__ b3, float* __restrict__ logits, int G)
{
    int g = blockIdx.x * blockDim.x + threadIdx.x;
    if (g >= G) return;
    float z0 = b3[0], z1 = b3[1];
    #pragma unroll 8
    for (int k = 0; k < FEAT; ++k) {
        float v = ge[g * FEAT + k];
        z0 = fmaf(v, w3[k * 2 + 0], z0);
        z1 = fmaf(v, w3[k * 2 + 1], z1);
    }
    float m = fmaxf(z0, z1);
    float lse = m + logf(expf(z0 - m) + expf(z1 - m));
    logits[g * 2 + 0] = z0 - lse;
    logits[g * 2 + 1] = z1 - lse;
}

// ------------------------------------------------------------------------- launcher

extern "C" void kernel_launch(void* const* d_in, const int* in_sizes, int n_in,
                              void* d_out, int out_size, void* d_ws, size_t ws_size,
                              hipStream_t stream)
{
    (void)n_in; (void)ws_size;
    const float* x      = (const float*)d_in[0];
    const int*   ei     = (const int*)  d_in[1];
    const int*   batch  = (const int*)  d_in[2];
    const float* conv_w = (const float*)d_in[3];
    // d_in[4] = conv_b: cancels exactly under training-mode BN (constant shift)
    const float* bn_g   = (const float*)d_in[5];
    const float* bn_b   = (const float*)d_in[6];
    const float* fc1_w  = (const float*)d_in[7];
    const float* fc1_b  = (const float*)d_in[8];
    const float* fc2_w  = (const float*)d_in[9];
    const float* fc2_b  = (const float*)d_in[10];
    const float* fc3_w  = (const float*)d_in[11];
    const float* fc3_b  = (const float*)d_in[12];

    const int N = in_sizes[0] / FEAT;
    const int E = in_sizes[1] / 2;
    const int G = (out_size - N * FEAT) / (FEAT + 2);
    const int L = in_sizes[3] / (FEAT * FEAT);

    // d_out layout: node_embeddings [N,64] | graph_embedding [G,64] | logits [G,2]
    float* node_emb = (float*)d_out;
    float* ge       = node_emb + (size_t)N * FEAT;
    float* logits   = ge + (size_t)G * FEAT;

    // workspace: agg [N,64] | dinv [N] | bn stats [128]   (~26 MB)
    float* agg   = (float*)d_ws;
    float* dinv  = agg + (size_t)N * FEAT;
    float* stats = dinv + N;

    const int TB = 256;
    auto cdiv = [](int a, int b) { return (a + b - 1) / b; };

    // symmetric GCN normalization (degree incl. self loop -> rsqrt)
    fill_f32<<<cdiv(N, TB), TB, 0, stream>>>(dinv, 1.0f, N);
    deg_accum<<<cdiv(E, TB), TB, 0, stream>>>(ei, dinv, E);
    rsqrt_inplace<<<cdiv(N, TB), TB, 0, stream>>>(dinv, N);

    const int gemmBlocks = cdiv(N, 128);       // 4 waves x 32 rows per block
    const int NF = N * FEAT;

    const float* hin = x;
    for (int l = 0; l < L; ++l) {
        // linear (WMMA) -> stage in node_emb region of d_out
        gemm64_wmma<false, false><<<gemmBlocks, 128, 0, stream>>>(
            hin, conv_w + (size_t)l * FEAT * FEAT, nullptr, node_emb, nullptr, N);
        // aggregate
        fill_f32<<<cdiv(NF, TB), TB, 0, stream>>>(agg, 0.0f, NF);
        int msgThreads = (E + N) * 4;
        scatter_msgs<<<cdiv(msgThreads, TB), TB, 0, stream>>>(node_emb, ei, dinv, agg, E, N);
        // batchnorm + relu
        fill_f32<<<1, 128, 0, stream>>>(stats, 0.0f, 128);
        bn_stats<<<cdiv(N, 256), 256, 0, stream>>>(agg, stats, N);
        float* bnout = (l == L - 1) ? node_emb : agg;  // last layer finalizes node_embeddings
        bn_apply_relu<<<cdiv(NF, TB), TB, 0, stream>>>(agg, stats,
                                                       bn_g + l * FEAT, bn_b + l * FEAT,
                                                       bnout, N);
        hin = agg;
    }

    // fc1 (WMMA) -> out1 in agg
    gemm64_wmma<true, false><<<gemmBlocks, 128, 0, stream>>>(node_emb, fc1_w, fc1_b,
                                                             agg, nullptr, N);
    // fc2 (WMMA) fused with segment-max pooling into graph_embedding
    fill_f32<<<cdiv(G * FEAT, TB), TB, 0, stream>>>(ge, -FLT_MAX, G * FEAT);
    gemm64_wmma<true, true><<<gemmBlocks, 128, 0, stream>>>(agg, fc2_w, fc2_b, ge, batch, N);
    // classifier head + log_softmax
    head_logsoftmax<<<1, 128, 0, stream>>>(ge, fc3_w, fc3_b, logits, G);
}
